// SSMImputer_68564857913685
// MI455X (gfx1250) — matrix operand order repaired and checked by reading, not compiled, not run on hardware
//
#include <hip/hip_runtime.h>
#include <math.h>

// ---------------------------------------------------------------------------
// Model constants (from the reference)
// ---------------------------------------------------------------------------
#define SEQ_L   512
#define NBATCH  4
#define T_TOK   (SEQ_L * NBATCH)   // 2048 tokens
#define D_STATE 16

typedef __attribute__((ext_vector_type(16))) _Float16 v16h;
typedef __attribute__((ext_vector_type(8)))  float    v8f;
typedef __attribute__((ext_vector_type(4)))  float    f32x4;

// ---------------------------------------------------------------------------
// WMMA GEMM:  out[M,N] = gamma * ( alpha * (A[M,K] @ W[N,K]^T) + bias[N] + resid[M,N] )
// One wave computes a 16x64 tile (4 accumulators, A fragment reused 4x).
// f32 -> f16 convert on load, f32 accumulate.
// Requirements (guaranteed by this model): M % 16 == 0, K % 16 == 0.
// Main loop is branch-free with b128 loads; optional 16-wide K tail uses
// constant zeros for the upper 16 k of A and B (B rows address-clamped).
// N overhang handled by clamping the W row index (garbage columns are never
// stored). trans!=0 stores out[(m/L)*N*L + n*L + (m%L)] (token-major -> B,C,L).
// ---------------------------------------------------------------------------
__global__ __launch_bounds__(32) void k_gemm(
    const float* __restrict__ A, const float* __restrict__ W,
    const float* __restrict__ bias, const float* __restrict__ resid,
    float* __restrict__ out, int M, int N, int K,
    float alpha, float gamma, int trans, int Lout)
{
    const int lane = threadIdx.x;          // 0..31
    const int mt   = blockIdx.x;           // M / 16 tiles
    const int nt0  = blockIdx.y * 4;       // first of 4 N tiles
    const int half = lane >> 4;            // 0: lanes 0-15, 1: lanes 16-31
    const int lm   = lane & 15;
    const int mrow = mt * 16 + lm;

    v8f acc[4];
#pragma unroll
    for (int t = 0; t < 4; t++)
#pragma unroll
        for (int r = 0; r < 8; r++) acc[t][r] = 0.0f;

    const float* Arow = A + (size_t)mrow * K;

    // Clamped W row indices (keeps all loads in-bounds without branches)
    int ncl[4];
#pragma unroll
    for (int t = 0; t < 4; t++) {
        int n = (nt0 + t) * 16 + lm;
        ncl[t] = (n < N) ? n : (N - 1);
    }

    const int K32 = K & ~31;
    for (int k0 = 0; k0 < K32; k0 += 32) {
        // A fragment: lane needs A[mrow][k0+half*8 .. +7] and [k0+16+half*8 .. +7]
        const float* ap = Arow + k0 + (half << 3);
        f32x4 a0 = *(const f32x4*)(ap);
        f32x4 a1 = *(const f32x4*)(ap + 4);
        f32x4 a2 = *(const f32x4*)(ap + 16);
        f32x4 a3 = *(const f32x4*)(ap + 20);
        v16h a;
#pragma unroll
        for (int i = 0; i < 4; i++) {
            a[i]      = (_Float16)a0[i];
            a[4 + i]  = (_Float16)a1[i];
            a[8 + i]  = (_Float16)a2[i];
            a[12 + i] = (_Float16)a3[i];
        }
#pragma unroll
        for (int t = 0; t < 4; t++) {
            // B fragment: lane needs W[n][k0+half*8 .. +7] and [k0+16+half*8 .. +7] (B = W^T)
            const float* bp = W + (size_t)ncl[t] * K + k0 + (half << 3);
            f32x4 b0 = *(const f32x4*)(bp);
            f32x4 b1 = *(const f32x4*)(bp + 4);
            f32x4 b2 = *(const f32x4*)(bp + 16);
            f32x4 b3 = *(const f32x4*)(bp + 20);
            v16h bm;
#pragma unroll
            for (int i = 0; i < 4; i++) {
                bm[i]      = (_Float16)b0[i];
                bm[4 + i]  = (_Float16)b1[i];
                bm[8 + i]  = (_Float16)b2[i];
                bm[12 + i] = (_Float16)b3[i];
            }
            acc[t] = __builtin_amdgcn_wmma_f32_16x16x32_f16(
                false, a, false, bm, (short)0, acc[t], false, false);
        }
    }

    if (K & 16) {   // 16-wide K tail (K % 32 == 16)
        const int k0 = K32;
        const float* ap = Arow + k0 + (half << 3);   // k0+half*8+7 <= K-1: in bounds
        f32x4 a0 = *(const f32x4*)(ap);
        f32x4 a1 = *(const f32x4*)(ap + 4);
        v16h a;
#pragma unroll
        for (int i = 0; i < 4; i++) {
            a[i]      = (_Float16)a0[i];
            a[4 + i]  = (_Float16)a1[i];
            a[8 + i]  = (_Float16)0.0f;   // K >= k0+16 region: zero
            a[12 + i] = (_Float16)0.0f;
        }

#pragma unroll
        for (int t = 0; t < 4; t++) {
            const float* bp = W + (size_t)ncl[t] * K + k0 + (half << 3);  // clamped addr, in bounds
            f32x4 b0 = *(const f32x4*)(bp);
            f32x4 b1 = *(const f32x4*)(bp + 4);
            v16h bm;
#pragma unroll
            for (int i = 0; i < 4; i++) {
                bm[i]      = (_Float16)b0[i];
                bm[4 + i]  = (_Float16)b1[i];
                bm[8 + i]  = (_Float16)0.0f;
                bm[12 + i] = (_Float16)0.0f;
            }
            acc[t] = __builtin_amdgcn_wmma_f32_16x16x32_f16(
                false, a, false, bm, (short)0, acc[t], false, false);
        }
    }

    // Epilogue. C/D layout: VGPR r -> m = r + 8*half ; lane&15 -> n.
#pragma unroll
    for (int t = 0; t < 4; t++) {
        int n = (nt0 + t) * 16 + lm;
        if (n >= N) continue;
        float bv = bias ? bias[n] : 0.0f;
#pragma unroll
        for (int r = 0; r < 8; r++) {
            int m = mt * 16 + (half << 3) + r;
            float v = alpha * acc[t][r] + bv;
            if (resid) v += resid[(size_t)m * N + n];
            v *= gamma;
            if (!trans) {
                out[(size_t)m * N + n] = v;
            } else {
                int b = m / Lout, l = m - b * Lout;
                out[((size_t)b * N + n) * Lout + l] = v;
            }
        }
    }
}

// ---------------------------------------------------------------------------
// RMSNorm over last dim (one thread per token; d <= 256)
// ---------------------------------------------------------------------------
__global__ void k_rmsnorm(const float* __restrict__ x, const float* __restrict__ w,
                          float* __restrict__ o, int d, int M)
{
    int m = blockIdx.x * blockDim.x + threadIdx.x;
    if (m >= M) return;
    const float* row = x + (size_t)m * d;
    float ss = 0.0f;
    for (int c = 0; c < d; c += 4) {
        f32x4 v = *(const f32x4*)(row + c);
        ss += v[0]*v[0] + v[1]*v[1] + v[2]*v[2] + v[3]*v[3];
    }
    float r = rsqrtf(ss / (float)d + 1e-5f);
    float* orow = o + (size_t)m * d;
    for (int c = 0; c < d; c += 4) {
        f32x4 v = *(const f32x4*)(row + c);
        f32x4 wv = *(const f32x4*)(w + c);
        f32x4 ov;
        ov[0] = v[0]*r*wv[0]; ov[1] = v[1]*r*wv[1];
        ov[2] = v[2]*r*wv[2]; ov[3] = v[3]*r*wv[3];
        *(f32x4*)(orow + c) = ov;
    }
}

// ---------------------------------------------------------------------------
// Causal depthwise conv (k=4) + bias + SiLU (branch-free boundary handling).
// xz is (T, 2*ed); channel e of the first half. Output xc (T, ed).
// ---------------------------------------------------------------------------
__global__ void k_dwconv_silu(const float* __restrict__ xz, const float* __restrict__ cw,
                              const float* __restrict__ cb, float* __restrict__ xc,
                              int ed, int total)
{
    int i = blockIdx.x * blockDim.x + threadIdx.x;
    if (i >= total) return;
    int e = i % ed, m = i / ed;
    int l = m & (SEQ_L - 1);
    int stride = 2 * ed;
    float s = cb[e];
#pragma unroll
    for (int j = 0; j < 4; j++) {
        int dl = l - 3 + j;
        int mm = (dl >= 0) ? (m - 3 + j) : m;          // clamp address, stay in-bounds
        float v = xz[(size_t)mm * stride + e];
        s += (dl >= 0) ? cw[e * 4 + j] * v : 0.0f;     // cndmask, no branch
    }
    xc[i] = s * (1.0f / (1.0f + expf(-s)));
}

// ---------------------------------------------------------------------------
// Fused selective scan: dt-projection + softplus + exp(dA) recurrence +
// C-contraction + D skip + silu(z) gate.  One thread per (batch, channel).
// Per-step dbc row (dt_rank + 2*16 <= 48 floats) broadcast via LDS.
// ---------------------------------------------------------------------------
__global__ __launch_bounds__(128) void k_scan(
    const float* __restrict__ dbc, const float* __restrict__ dtw,
    const float* __restrict__ dtb, const float* __restrict__ alog,
    const float* __restrict__ Dp,  const float* __restrict__ xc,
    const float* __restrict__ xz,  float* __restrict__ y,
    int ed, int dtr)
{
    __shared__ float sdbc[48];
    const int b = blockIdx.x;
    const int e = blockIdx.y * 128 + threadIdx.x;
    const bool act = (e < ed);
    const int nd = dtr + 2 * D_STATE;
    const int zstride = 2 * ed;

    float wreg[16], h[D_STATE], Aa[D_STATE];
    float db = 0.0f, Dv = 0.0f;
    if (act) {
#pragma unroll
        for (int r = 0; r < 16; r++) wreg[r] = (r < dtr) ? dtw[(size_t)e * dtr + r] : 0.0f;
        db = dtb[e];
        Dv = Dp[e];
#pragma unroll
        for (int n = 0; n < D_STATE; n++) {
            Aa[n] = -expf(alog[(size_t)e * D_STATE + n]);
            h[n]  = 0.0f;
        }
    }

    for (int l = 0; l < SEQ_L; l++) {
        int m = b * SEQ_L + l;
        if (threadIdx.x < (unsigned)nd) sdbc[threadIdx.x] = dbc[(size_t)m * nd + threadIdx.x];
        __syncthreads();
        if (act) {
            float t = db;
            for (int r = 0; r < dtr; r++) t += sdbc[r] * wreg[r];
            float delta = (t > 20.0f) ? t : log1pf(expf(t));
            float xv = xc[(size_t)m * ed + e];
            float accv = 0.0f;
#pragma unroll
            for (int n = 0; n < D_STATE; n++) {
                h[n] = expf(delta * Aa[n]) * h[n] + delta * sdbc[dtr + n] * xv;
                accv += h[n] * sdbc[dtr + D_STATE + n];
            }
            accv += xv * Dv;
            float zv = xz[(size_t)m * zstride + ed + e];
            accv *= zv * (1.0f / (1.0f + expf(-zv)));
            y[(size_t)m * ed + e] = accv;
        }
        __syncthreads();
    }
}

// ---------------------------------------------------------------------------
// Small elementwise / layout kernels
// ---------------------------------------------------------------------------
__global__ void k_add(const float* __restrict__ a, const float* __restrict__ b,
                      float* __restrict__ o, int n)
{
    int i = blockIdx.x * blockDim.x + threadIdx.x;
    if (i < n) o[i] = a[i] + b[i];
}

__global__ void k_add_bcast_de(float* __restrict__ h, const float* __restrict__ de, int n)
{
    int i = blockIdx.x * blockDim.x + threadIdx.x;   // i = m*128 + c
    if (i >= n) return;
    int c = i & 127;
    int b = (i >> 7) / SEQ_L;
    h[i] += de[b * 128 + c];
}

__global__ void k_gate(const float* __restrict__ h, float* __restrict__ o, int n)
{
    int i = blockIdx.x * blockDim.x + threadIdx.x;   // i = m*128 + c
    if (i >= n) return;
    int m = i >> 7, c = i & 127;
    float a = h[(size_t)m * 256 + c];
    float g = h[(size_t)m * 256 + 128 + c];
    o[i] = tanhf(a) * (1.0f / (1.0f + expf(-g)));
}

__global__ void k_bcl_to_blc(const float* __restrict__ in, float* __restrict__ out,
                             int C, int n)
{
    int i = blockIdx.x * blockDim.x + threadIdx.x;   // i = m*C + c
    if (i >= n) return;
    int c = i % C, m = i / C;
    int b = m / SEQ_L, l = m - b * SEQ_L;
    out[i] = in[((size_t)b * C + c) * SEQ_L + l];
}

__global__ void k_build_cond2(const float* __restrict__ cond, const float* __restrict__ msk,
                              float* __restrict__ out, int n)
{
    int i = blockIdx.x * blockDim.x + threadIdx.x;   // i = m*128 + c
    if (i >= n) return;
    int c = i & 127, m = i >> 7;
    int b = m / SEQ_L, l = m - b * SEQ_L;
    float v;
    if (c < 64) {
        size_t idx = ((size_t)b * 64 + c) * SEQ_L + l;
        v = cond[idx] * msk[idx];
    } else {
        size_t idx = ((size_t)b * 64 + (c - 64)) * SEQ_L + l;
        v = msk[idx];
    }
    out[i] = v;
}

// diffusion embedding + diff_proj:  de[b,o] = sum_k demb[b,k]*W[o,k] + bias[o]
__global__ __launch_bounds__(128) void k_diff_embed(
    const float* __restrict__ step, const float* __restrict__ W,
    const float* __restrict__ bias, float* __restrict__ de)
{
    __shared__ float sd[NBATCH][128];
    int t = threadIdx.x;                     // 0..127
    float cc = -logf(10000.0f) / 63.0f;      // half-1 = 63
    for (int b = 0; b < NBATCH; b++) {
        float s = step[b];
        float v = (t < 64) ? sinf(s * expf(cc * (float)t))
                           : cosf(s * expf(cc * (float)(t - 64)));
        sd[b][t] = v;
    }
    __syncthreads();
    for (int b = 0; b < NBATCH; b++) {
        float acc = bias[t];
        for (int k = 0; k < 128; k++) acc += sd[b][k] * W[t * 128 + k];
        de[b * 128 + t] = acc;
    }
}

// ---------------------------------------------------------------------------
// Host-side orchestration
// ---------------------------------------------------------------------------
struct Scratch {
    float *nrm, *xz, *xc, *dbc, *yb, *r0, *r1, *r2;
};

static void launch_gemm(const float* A, const float* W, const float* bias,
                        const float* resid, float* out, int M, int N, int K,
                        float alpha, float gamma, int trans, hipStream_t st)
{
    dim3 g((unsigned)(M / 16), (unsigned)((N + 63) / 64), 1);
    k_gemm<<<g, dim3(32), 0, st>>>(A, W, bias, resid, out, M, N, K,
                                   alpha, gamma, trans, SEQ_L);
}

static void run_mamba_layer(void* const* d_in, int& ix, int d,
                            float*& cur, float*& alt, const Scratch& sb, hipStream_t st)
{
    const float* norm   = (const float*)d_in[ix++];
    const float* inproj = (const float*)d_in[ix++];
    const float* convw  = (const float*)d_in[ix++];
    const float* convb  = (const float*)d_in[ix++];
    const float* xproj  = (const float*)d_in[ix++];
    const float* dtw    = (const float*)d_in[ix++];
    const float* dtb    = (const float*)d_in[ix++];
    const float* alog   = (const float*)d_in[ix++];
    const float* Dp     = (const float*)d_in[ix++];
    const float* outp   = (const float*)d_in[ix++];
    int ed  = 2 * d;
    int dtr = (d + 15) / 16; if (dtr < 1) dtr = 1;

    k_rmsnorm<<<dim3((T_TOK + 255) / 256), dim3(256), 0, st>>>(cur, norm, sb.nrm, d, T_TOK);
    launch_gemm(sb.nrm, inproj, nullptr, nullptr, sb.xz, T_TOK, 4 * d, d, 1.f, 1.f, 0, st);
    int tot = T_TOK * ed;
    k_dwconv_silu<<<dim3((tot + 255) / 256), dim3(256), 0, st>>>(sb.xz, convw, convb, sb.xc, ed, tot);
    launch_gemm(sb.xc, xproj, nullptr, nullptr, sb.dbc, T_TOK, dtr + 2 * D_STATE, ed, 1.f, 1.f, 0, st);
    k_scan<<<dim3(NBATCH, (unsigned)((ed + 127) / 128)), dim3(128), 0, st>>>(
        sb.dbc, dtw, dtb, alog, Dp, sb.xc, sb.xz, sb.yb, ed, dtr);
    launch_gemm(sb.yb, outp, nullptr, cur, alt, T_TOK, d, ed, 1.f, 1.f, 0, st);
    float* t = cur; cur = alt; alt = t;
}

static void run_urm(void* const* d_in, int& ix, int d,
                    float*& cur, float*& alt, const Scratch& sb, hipStream_t st)
{
    float* rs[3] = { sb.r0, sb.r1, sb.r2 };
    int dc = d;
    for (int i = 0; i < 3; i++) {                               // downs
        hipMemcpyAsync(rs[i], cur, (size_t)T_TOK * dc * sizeof(float),
                       hipMemcpyDeviceToDevice, st);
        const float* w = (const float*)d_in[ix++];
        const float* b = (const float*)d_in[ix++];
        launch_gemm(cur, w, b, nullptr, alt, T_TOK, dc / 2, dc, 1.f, 1.f, 0, st);
        { float* t = cur; cur = alt; alt = t; }
        dc >>= 1;
        run_mamba_layer(d_in, ix, dc, cur, alt, sb, st);
    }
    for (int i = 0; i < 2; i++)                                 // mids
        run_mamba_layer(d_in, ix, dc, cur, alt, sb, st);
    for (int i = 0; i < 3; i++) {                               // ups
        const float* w = (const float*)d_in[ix++];
        const float* b = (const float*)d_in[ix++];
        launch_gemm(cur, w, b, nullptr, alt, T_TOK, dc * 2, dc, 1.f, 1.f, 0, st);
        { float* t = cur; cur = alt; alt = t; }
        dc <<= 1;
        run_mamba_layer(d_in, ix, dc, cur, alt, sb, st);
        int n = T_TOK * dc;
        k_add<<<dim3((n + 255) / 256), dim3(256), 0, st>>>(cur, rs[2 - i], cur, n);
    }
}

extern "C" void kernel_launch(void* const* d_in, const int* in_sizes, int n_in,
                              void* d_out, int out_size, void* d_ws, size_t ws_size,
                              hipStream_t stream)
{
    (void)in_sizes; (void)n_in; (void)out_size; (void)ws_size;
    const float* noise = (const float*)d_in[0];
    const float* cond  = (const float*)d_in[1];
    const float* maskp = (const float*)d_in[2];
    const float* step  = (const float*)d_in[3];
    int ix = 4;   // params follow in insertion (== execution) order

    // Workspace bump allocation (~36 MB of float32)
    float* base = (float*)d_ws;
    size_t off = 0;
    auto alloc = [&](size_t nf) { float* p = base + off; off += nf; return p; };

    float* cond2 = alloc((size_t)T_TOK * 128);
    float* xa    = alloc((size_t)T_TOK * 64);
    float* xb    = alloc((size_t)T_TOK * 64);
    float* skip  = alloc((size_t)T_TOK * 128);
    float* de    = alloc(NBATCH * 128);
    float* ha    = alloc((size_t)T_TOK * 256);
    float* hb    = alloc((size_t)T_TOK * 256);
    float* ca    = alloc((size_t)T_TOK * 256);
    float* cb    = alloc((size_t)T_TOK * 256);
    float* gate  = alloc((size_t)T_TOK * 128);
    Scratch sb;
    sb.nrm = alloc((size_t)T_TOK * 256);
    sb.xz  = alloc((size_t)T_TOK * 1024);
    sb.xc  = alloc((size_t)T_TOK * 512);
    sb.dbc = alloc((size_t)T_TOK * 64);
    sb.yb  = alloc((size_t)T_TOK * 512);
    sb.r0  = alloc((size_t)T_TOK * 256);
    sb.r1  = alloc((size_t)T_TOK * 128);
    sb.r2  = alloc((size_t)T_TOK * 64);

    // Inputs -> token-major (BLC)
    {
        int n = T_TOK * 128;
        k_build_cond2<<<dim3((n + 255) / 256), dim3(256), 0, stream>>>(cond, maskp, cond2, n);
        int n2 = T_TOK * 64;
        k_bcl_to_blc<<<dim3((n2 + 255) / 256), dim3(256), 0, stream>>>(noise, xa, 64, n2);
    }

    float* xcur = xa;
    float* xalt = xb;

    for (int blk = 0; blk < 2; blk++) {
        float* cur = ha;
        float* alt = hb;
        // input_proj (64 -> 128)
        {
            const float* w = (const float*)d_in[ix++];
            const float* b = (const float*)d_in[ix++];
            launch_gemm(xcur, w, b, nullptr, cur, T_TOK, 128, 64, 1.f, 1.f, 0, stream);
        }
        // input_ssm: 1 stack x 1 layer at d=128
        run_mamba_layer(d_in, ix, 128, cur, alt, sb, stream);
        // diff_proj + broadcast add
        {
            const float* w = (const float*)d_in[ix++];
            const float* b = (const float*)d_in[ix++];
            k_diff_embed<<<dim3(1), dim3(128), 0, stream>>>(step, w, b, de);
            int n = T_TOK * 128;
            k_add_bcast_de<<<dim3((n + 255) / 256), dim3(256), 0, stream>>>(cur, de, n);
        }
        // urm1 at d=128
        run_urm(d_in, ix, 128, cur, alt, sb, stream);
        // mid_projection (128 -> 256)
        {
            const float* w = (const float*)d_in[ix++];
            const float* b = (const float*)d_in[ix++];
            launch_gemm(cur, w, b, nullptr, alt, T_TOK, 256, 128, 1.f, 1.f, 0, stream);
            float* t = cur; cur = alt; alt = t;
        }
        // cond path: cond_conv (128 -> 256), then 1 stack x 2 layers at d=256
        float* ccur = ca;
        float* calt = cb;
        {
            const float* w = (const float*)d_in[ix++];
            const float* b = (const float*)d_in[ix++];
            launch_gemm(cond2, w, b, nullptr, ccur, T_TOK, 256, 128, 1.f, 1.f, 0, stream);
        }
        for (int j = 0; j < 2; j++)
            run_mamba_layer(d_in, ix, 256, ccur, calt, sb, stream);
        // h += c
        {
            int n = T_TOK * 256;
            k_add<<<dim3((n + 255) / 256), dim3(256), 0, stream>>>(cur, ccur, cur, n);
        }
        // urm2 at d=256
        run_urm(d_in, ix, 256, cur, alt, sb, stream);
        // gate: tanh(h[:,:128]) * sigmoid(h[:,128:])
        {
            int n = T_TOK * 128;
            k_gate<<<dim3((n + 255) / 256), dim3(256), 0, stream>>>(cur, gate, n);
        }
        // res_conv (128 -> 64): x_next = (x + res) * sqrt(0.5)
        {
            const float* w = (const float*)d_in[ix++];
            const float* b = (const float*)d_in[ix++];
            launch_gemm(gate, w, b, xcur, xalt, T_TOK, 64, 128, 1.f, 0.70710678f, 0, stream);
            float* t = xcur; xcur = xalt; xalt = t;
        }
        // skip_conv (128 -> 128): skip (+)= conv(gate)
        {
            const float* w = (const float*)d_in[ix++];
            const float* b = (const float*)d_in[ix++];
            launch_gemm(gate, w, b, blk ? skip : nullptr, skip, T_TOK, 128, 128,
                        1.f, 1.f, 0, stream);
        }
    }

    // out_proj1 on skip/sqrt(2), store transposed to (B, 64, L)
    {
        const float* w = (const float*)d_in[ix++];
        const float* b = (const float*)d_in[ix++];
        launch_gemm(skip, w, b, nullptr, (float*)d_out, T_TOK, 64, 128,
                    0.70710678f, 1.f, 1, stream);
    }
}
